// AxialAttention3DSpaceTime_17789754540099
// MI455X (gfx1250) — compile-verified, matrix operand order checked by reading, not run on hardware
//
#include <hip/hip_runtime.h>

// ---------------------------------------------------------------------------
// AxialAttention3DSpaceTime for MI455X (gfx1250, wave32, WMMA bf16)
// Problem constants from setup_inputs(): B=2, t=8, D=H=W=16, C=512,
// HEADS=8 (hd=64), RANK=16, LORA_SCALE=2.
// ---------------------------------------------------------------------------

typedef __attribute__((ext_vector_type(16))) __bf16 v16bf;
typedef __attribute__((ext_vector_type(8)))  float  v8f;

static_assert(sizeof(v16bf) == 32, "v16bf must be 32 bytes");

#define CB   2      // batch
#define CT   8      // time
#define CD   16
#define CH   16
#define CW   16
#define CC   512    // channels
#define CHEADS 8
#define CHD  64     // head dim
#define CN   (CD*CH*CW)        // 4096 spatial tokens
#define CROWS (CB*CT*CN)       // 65536 total tokens
#define MATN 16                // 4 axes * 4 projections
#define MATSZ (CC*CC)          // 262144

// LDS row pitch for 16x512 bf16 tiles (pad to break bank alignment)
#define XP 520
// V-transposed tile pitch (48B rows -> 16B aligned, conflict-free for 16 lanes)
#define VTP 24

union ABReg {
  uint4 q[2];
  v16bf v;
};

__device__ __forceinline__ unsigned short f2bf(float f) {
  union { float f; unsigned int u; } cv; cv.f = f;
  unsigned int r = cv.u + 0x7FFFu + ((cv.u >> 16) & 1u);  // RNE
  return (unsigned short)(r >> 16);
}

__device__ __forceinline__ v8f wmma_bf16(const ABReg& a, const ABReg& b, v8f c) {
  return __builtin_amdgcn_wmma_f32_16x16x32_bf16(
      false, a.v, false, b.v, (short)0, c, false, false);
}

// Token row index (into [CROWS]) for tile row r (0..15) of block bk on `axis`.
// Layout of x: (B, t, D, H, W, C) row-major; row index excludes the C dim.
__device__ __forceinline__ int token_row(int axis, int bk, int r) {
  if (axis == 0) {                     // time axis: 2 sequences of L=8 per tile
    int s  = bk * 2 + (r >> 3);        // seq id over (b, d, h, w)
    int ti = r & 7;
    int b  = s >> 12;                  // / 4096
    int sp = s & 4095;                 // d*256 + h*16 + w
    return ((b * CT + ti) << 12) + sp;
  }
  int bt = bk >> 8;                    // b*T + ti   (16 values)
  if (axis == 1) {                     // depth: seq over (b,t,h,w), token = d
    int hw = bk & 255;
    return (bt << 12) + (r << 8) + hw;
  }
  if (axis == 2) {                     // height: seq over (b,t,d,w), token = h
    int dw = bk & 255;
    int d = dw >> 4, w = dw & 15;
    return (bt << 12) + (d << 8) + (r << 4) + w;
  }
  // width: seq over (b,t,d,h), token = w
  int dh = bk & 255;
  return (bt << 12) + (dh << 4) + r;
}

// ---------------------------------------------------------------------------
// Fold LoRA into dense weights:  Weff[o,c] = W[o,c] + 2 * sum_r A[c,r]*Bl[r,o]
// stored bf16, [mat][o][c].
// ---------------------------------------------------------------------------
__global__ __launch_bounds__(256) void fold_weights_kernel(
    const float* __restrict__ W, const float* __restrict__ A,
    const float* __restrict__ Bl, unsigned short* __restrict__ Weff) {
  int idx = blockIdx.x * 256 + threadIdx.x;           // [0, 16*512*512)
  int mat = idx / MATSZ;
  int oc  = idx - mat * MATSZ;
  int o = oc >> 9, c = oc & 511;
  const float* Am  = A  + (size_t)mat * CC * 16 + c * 16;
  const float* Blm = Bl + (size_t)mat * 16 * CC + o;
  float lora = 0.f;
#pragma unroll
  for (int r = 0; r < 16; ++r) lora += Am[r] * Blm[(size_t)r * CC];
  float val = W[(size_t)mat * MATSZ + oc] + 2.0f * lora;
  Weff[(size_t)mat * MATSZ + (size_t)o * CC + c] = f2bf(val);
}

// ---------------------------------------------------------------------------
// One axial attention pass. 256 threads = 8 waves; one 16-token tile / block.
// axis==0 reads fp32 x, writes out = x + attn(x) and bf16 copy xpost.
// axis>0  reads bf16 xpost, does out += attn(xpost).
// ---------------------------------------------------------------------------
__global__ __launch_bounds__(256) void axial_pass_kernel(
    const float* __restrict__ x,
    const unsigned short* __restrict__ xpost_in,
    unsigned short* __restrict__ xpost_out,
    const unsigned short* __restrict__ Weff,
    const float* __restrict__ bias,
    float* __restrict__ out,
    int axis) {
  __shared__ unsigned short sX[16 * XP];      // input tile, reused as Y tile
  __shared__ unsigned short sQ[16 * XP];
  __shared__ unsigned short sK[16 * XP];
  __shared__ unsigned short sVt[CC * VTP];    // V transposed: [channel][token]
  __shared__ unsigned short sAttn[8 * 16 * 16];

  const int tid  = threadIdx.x;
  const int wv   = tid >> 5;
  const int lane = tid & 31;
  const int col  = lane & 15;                 // output-col / token within tile
  const int kb   = (lane >> 4) ? 8 : 0;      // K-chunk base per A/B layout
  const int rb   = (lane >> 4) << 3;         // C/D row base (0 or 8)
  const int bk   = blockIdx.x;

  // ---- Phase 1: stage X tile (16 x 512) into LDS as bf16 -------------------
  {
    int r  = tid >> 4;
    int c0 = (tid & 15) * 32;
    int grow = token_row(axis, bk, r);
    unsigned short* dst = sX + r * XP + c0;
    if (axis == 0) {
      const float* src = x + (size_t)grow * CC + c0;
#pragma unroll
      for (int j = 0; j < 8; ++j) {
        float4 f = *(const float4*)(src + j * 4);
        dst[j*4+0] = f2bf(f.x); dst[j*4+1] = f2bf(f.y);
        dst[j*4+2] = f2bf(f.z); dst[j*4+3] = f2bf(f.w);
      }
    } else {
      const unsigned short* src = xpost_in + (size_t)grow * CC + c0;
#pragma unroll
      for (int j = 0; j < 4; ++j)
        *(uint4*)(dst + j * 8) = *(const uint4*)(src + j * 8);
    }
  }
  __syncthreads();

  // ---- Phase 2: Q, K, V projections (16 x 512 each) ------------------------
#pragma unroll
  for (int m = 0; m < 3; ++m) {
    const unsigned short* Wm = Weff + (size_t)(axis * 4 + m) * MATSZ;
    for (int tt = wv; tt < 32; tt += 8) {
      int o = tt * 16 + col;
      v8f acc = {0.f,0.f,0.f,0.f,0.f,0.f,0.f,0.f};
      const unsigned short* arow = sX + col * XP;
      const unsigned short* brow = Wm + (size_t)o * CC;
#pragma unroll 4
      for (int ks = 0; ks < 16; ++ks) {
        ABReg a, b;
        a.q[0] = *(const uint4*)(arow + ks * 32 + kb);
        a.q[1] = *(const uint4*)(arow + ks * 32 + kb + 16);
        b.q[0] = *(const uint4*)(brow + ks * 32 + kb);
        b.q[1] = *(const uint4*)(brow + ks * 32 + kb + 16);
        acc = wmma_bf16(a, b, acc);
      }
      float bv = bias[(axis * 4 + m) * CC + o];
#pragma unroll
      for (int vi = 0; vi < 8; ++vi) {
        int row = vi + rb;
        unsigned short bfv = f2bf(acc[vi] + bv);
        if (m == 0)      sQ[row * XP + o] = bfv;
        else if (m == 1) sK[row * XP + o] = bfv;
        else             sVt[o * VTP + row] = bfv;   // transposed
      }
    }
  }
  __syncthreads();

  // ---- Phase 3: attention, one head per wave ------------------------------
  {
    const int head = wv;
    v8f sc = {0.f,0.f,0.f,0.f,0.f,0.f,0.f,0.f};
#pragma unroll
    for (int ks = 0; ks < 2; ++ks) {          // K = head-dim 64 = 2 x 32
      ABReg a, b;
      const unsigned short* qa = sQ + col * XP + head * CHD + ks * 32 + kb;
      const unsigned short* kbp = sK + col * XP + head * CHD + ks * 32 + kb;
      a.q[0] = *(const uint4*)(qa);
      a.q[1] = *(const uint4*)(qa + 16);
      b.q[0] = *(const uint4*)(kbp);
      b.q[1] = *(const uint4*)(kbp + 16);
      sc = wmma_bf16(a, b, sc);
    }
    const float scale = 0.125f;               // 1/sqrt(64)
    unsigned short* at = sAttn + wv * 256;
#pragma unroll
    for (int vi = 0; vi < 8; ++vi) {
      int row = vi + rb;
      bool valid = (axis != 0) || ((row >> 3) == (col >> 3));  // block-diag mask
      float s = valid ? sc[vi] * scale : -1e30f;
      float mx = s;
#pragma unroll
      for (int off = 8; off >= 1; off >>= 1)
        mx = fmaxf(mx, __shfl_xor(mx, off, 32));
      float e = __expf(s - mx);
      float sm = e;
#pragma unroll
      for (int off = 8; off >= 1; off >>= 1)
        sm += __shfl_xor(sm, off, 32);
      at[row * 16 + col] = f2bf(e / sm);
    }
    // y = attn @ V  (K=16 tokens, zero-padded to 32)
    ABReg a;
    a.q[0] = *(const uint4*)(at + col * 16 + kb);
    a.q[1] = make_uint4(0u, 0u, 0u, 0u);
    unsigned short* sY = sX;                  // alias: X no longer needed
#pragma unroll
    for (int nt = 0; nt < 4; ++nt) {
      int ch = head * CHD + nt * 16 + col;
      ABReg b;
      b.q[0] = *(const uint4*)(sVt + ch * VTP + kb);
      b.q[1] = make_uint4(0u, 0u, 0u, 0u);
      v8f y = {0.f,0.f,0.f,0.f,0.f,0.f,0.f,0.f};
      y = wmma_bf16(a, b, y);
#pragma unroll
      for (int vi = 0; vi < 8; ++vi)
        sY[(vi + rb) * XP + ch] = f2bf(y[vi]);
    }
  }
  __syncthreads();

  // ---- Phase 4: output projection + global accumulate ----------------------
  {
    const unsigned short* Wm = Weff + (size_t)(axis * 4 + 3) * MATSZ;
    const unsigned short* sY = sX;
    for (int tt = wv; tt < 32; tt += 8) {
      int o = tt * 16 + col;
      v8f acc = {0.f,0.f,0.f,0.f,0.f,0.f,0.f,0.f};
      const unsigned short* arow = sY + col * XP;
      const unsigned short* brow = Wm + (size_t)o * CC;
#pragma unroll 4
      for (int ks = 0; ks < 16; ++ks) {
        ABReg a, b;
        a.q[0] = *(const uint4*)(arow + ks * 32 + kb);
        a.q[1] = *(const uint4*)(arow + ks * 32 + kb + 16);
        b.q[0] = *(const uint4*)(brow + ks * 32 + kb);
        b.q[1] = *(const uint4*)(brow + ks * 32 + kb + 16);
        acc = wmma_bf16(a, b, acc);
      }
      float bv = bias[(axis * 4 + 3) * CC + o];
#pragma unroll
      for (int vi = 0; vi < 8; ++vi) {
        int row = vi + rb;
        int grow = token_row(axis, bk, row);
        size_t gi = (size_t)grow * CC + o;
        float val = acc[vi] + bv;
        if (axis == 0) {
          float tot = x[gi] + val;             // x6 = x + xt   (t > 1)
          out[gi] = tot;
          xpost_out[gi] = f2bf(tot);
        } else {
          out[gi] += val;                      // out = x6 + xd + xh + xw
        }
      }
    }
  }
}

// ---------------------------------------------------------------------------
extern "C" void kernel_launch(void* const* d_in, const int* in_sizes, int n_in,
                              void* d_out, int out_size, void* d_ws, size_t ws_size,
                              hipStream_t stream) {
  const float* x  = (const float*)d_in[0];   // (2,8,4096,512) fp32
  const float* W  = (const float*)d_in[1];   // (4,4,512,512)
  const float* b  = (const float*)d_in[2];   // (4,4,512)
  const float* A  = (const float*)d_in[3];   // (4,4,512,16)
  const float* Bl = (const float*)d_in[4];   // (4,4,16,512)
  float* out = (float*)d_out;

  unsigned short* Weff  = (unsigned short*)d_ws;                      // 8 MB
  unsigned short* xpost = (unsigned short*)((char*)d_ws +
                          (size_t)MATN * MATSZ * sizeof(unsigned short)); // 64 MB

  fold_weights_kernel<<<(MATN * MATSZ) / 256, 256, 0, stream>>>(W, A, Bl, Weff);

  for (int axis = 0; axis < 4; ++axis) {
    axial_pass_kernel<<<CROWS / 16, 256, 0, stream>>>(
        x, xpost, xpost, Weff, b, out, axis);
  }
}
